// NSAAttention_49486613184733
// MI455X (gfx1250) — matrix operand order repaired and checked by reading, not compiled, not run on hardware
//
#include <hip/hip_runtime.h>
#include <math.h>

// ---------------- problem constants ----------------
#define S_LEN   2048
#define DIM_    2048
#define NHEADS  16
#define G_GRP   4
#define H_PER_G 4
#define DK_     128
#define L_CMP   32
#define D_STR   16
#define LSEL_   64
#define NSEL_   16
#define W_WIN   512
#define GH_     64
#define NC_     127
#define NCP_    128
#define NB_     32
#define KV_COLS (G_GRP * DK_)   // 512
#define Q_COLS  (NHEADS * DK_)  // 2048
#define SCALE_  0.08838834764831845f
#define NEGL    (-1e30f)

typedef __attribute__((ext_vector_type(16))) _Float16 v16h;
typedef __attribute__((ext_vector_type(8)))  float    v8f;

__device__ __forceinline__ v8f wmma16(v16h a, v16h b, v8f c) {
  // D = A(16x32 f16) * B(32x16 f16) + C(16x16 f32)
  return __builtin_amdgcn_wmma_f32_16x16x32_f16(false, a, false, b, (short)0, c,
                                                false, false);
}

// xor-shuffle via ds_swizzle (group-of-32 mode: and=0x1f, or=0, xor=m)
#define SWZ_XOR_F(v, m) \
  __int_as_float(__builtin_amdgcn_ds_swizzle(__float_as_int(v), (0x1f | ((m) << 10))))

__device__ __forceinline__ float redmax16(float v) {
  v = fmaxf(v, SWZ_XOR_F(v, 1));
  v = fmaxf(v, SWZ_XOR_F(v, 2));
  v = fmaxf(v, SWZ_XOR_F(v, 4));
  v = fmaxf(v, SWZ_XOR_F(v, 8));
  return v;
}
__device__ __forceinline__ float redsum16(float v) {
  v += SWZ_XOR_F(v, 1);
  v += SWZ_XOR_F(v, 2);
  v += SWZ_XOR_F(v, 4);
  v += SWZ_XOR_F(v, 8);
  return v;
}

// ---------------------------------------------------------------------------
// GEMM C = A[M,K](f32) @ B[K,N](f32) via f16 WMMA, fp32 accumulate.
// OT   = output element type (float or _Float16)
// TRO  = false: C[M,N] row-major;  true: C stored transposed as C_t[N,M]
// Block = 128 threads (4 waves), tile 64x64, K step 32. M%64==N%64==K%32==0.
// ---------------------------------------------------------------------------
template <typename OT, bool TRO>
__global__ __launch_bounds__(128) void gemm_f16_kernel(
    const float* __restrict__ A, const float* __restrict__ B,
    OT* __restrict__ C, int M, int N, int K)
{
  __shared__ _Float16 As[64 * 32];   // [m][k]
  __shared__ _Float16 Bs[64 * 32];   // [n][k] (transposed on store)
  const int t    = threadIdx.x;
  const int lane = t & 31;
  const int wave = t >> 5;
  const int m0 = blockIdx.y * 64;
  const int n0 = blockIdx.x * 64;
  const int r     = lane & 15;
  const int kbA   = (lane < 16) ? 0 : 16;
  const int kbB   = (lane < 16) ? 0 : 16;
  const int rbase = (lane < 16) ? 0 : 8;

  v8f acc[4];
#pragma unroll
  for (int nt = 0; nt < 4; ++nt)
#pragma unroll
    for (int i = 0; i < 8; ++i) acc[nt][i] = 0.0f;

  for (int k0 = 0; k0 < K; k0 += 32) {
#pragma unroll
    for (int p = 0; p < 4; ++p) {
      const int idx  = (p * 128 + t) * 4;      // 0..2047
      const int arow = idx >> 5, acol = idx & 31;
      const float4 va = *(const float4*)(A + (size_t)(m0 + arow) * K + k0 + acol);
      As[arow * 32 + acol + 0] = (_Float16)va.x;
      As[arow * 32 + acol + 1] = (_Float16)va.y;
      As[arow * 32 + acol + 2] = (_Float16)va.z;
      As[arow * 32 + acol + 3] = (_Float16)va.w;
      const int krow = idx >> 6, ncol = idx & 63;
      const float4 vb = *(const float4*)(B + (size_t)(k0 + krow) * N + n0 + ncol);
      Bs[(ncol + 0) * 32 + krow] = (_Float16)vb.x;
      Bs[(ncol + 1) * 32 + krow] = (_Float16)vb.y;
      Bs[(ncol + 2) * 32 + krow] = (_Float16)vb.z;
      Bs[(ncol + 3) * 32 + krow] = (_Float16)vb.w;
    }
    __syncthreads();
    v16h af;
    const int arow2 = wave * 16 + r;
#pragma unroll
    for (int i = 0; i < 8; ++i) {
      af[i]     = As[arow2 * 32 + kbA + i];
      af[8 + i] = As[arow2 * 32 + kbA + 8 + i];
    }
#pragma unroll
    for (int nt = 0; nt < 4; ++nt) {
      v16h bf;
#pragma unroll
      for (int i = 0; i < 16; ++i) bf[i] = Bs[(nt * 16 + r) * 32 + kbB + i];
      acc[nt] = wmma16(af, bf, acc[nt]);
    }
    __syncthreads();
  }
  if (TRO) {
    // transposed store: per lane 8 contiguous elements along M
#pragma unroll
    for (int nt = 0; nt < 4; ++nt)
#pragma unroll
      for (int rr = 0; rr < 8; ++rr)
        C[(size_t)(n0 + nt * 16 + r) * M + m0 + wave * 16 + rbase + rr] =
            (OT)acc[nt][rr];
  } else {
#pragma unroll
    for (int nt = 0; nt < 4; ++nt)
#pragma unroll
      for (int rr = 0; rr < 8; ++rr)
        C[(size_t)(m0 + wave * 16 + rbase + rr) * N + n0 + nt * 16 + r] =
            (OT)acc[nt][rr];
  }
}

// ---------------------------------------------------------------------------
// RoPE in place on f16 X laid out (S, nheads, 128); pos = s, half = 64.
// ---------------------------------------------------------------------------
__global__ void rope_kernel(_Float16* __restrict__ X, int nheads)
{
  const int idx = blockIdx.x * blockDim.x + threadIdx.x;
  const int j  = idx & 63;
  const int hh = (idx >> 6) % nheads;
  const int s  = idx / (64 * nheads);
  if (s >= S_LEN) return;
  const float inv = __powf(10000.0f, -(float)j * (1.0f / 64.0f));
  const float ang = (float)s * inv;
  const float c = __cosf(ang), sn = __sinf(ang);
  _Float16* p = X + ((size_t)s * nheads + hh) * 128;
  const float x1 = (float)p[j], x2 = (float)p[j + 64];
  p[j]      = (_Float16)(x1 * c - x2 * sn);
  p[j + 64] = (_Float16)(x2 * c + x1 * sn);
}

// ---------------------------------------------------------------------------
// Compressed K/V: mean over L=32 tokens, stride 16.
// Kcr:(S,G,DK) f16 -> Kcmp:(G,128,DK) f16 (row 127 zero)
// Vt:(G,DK,S) f16  -> Vcmpt:(G,DK,128) f16 (col 127 zero)
// ---------------------------------------------------------------------------
__global__ __launch_bounds__(128) void compress_kernel(
    const _Float16* __restrict__ Kcr, const _Float16* __restrict__ Vt,
    _Float16* __restrict__ Kcmp, _Float16* __restrict__ Vcmpt)
{
  const int d = threadIdx.x;
  const int c = blockIdx.x % NCP_;
  const int g = blockIdx.x / NCP_;
  float ks = 0.0f, vs = 0.0f;
  if (c < NC_) {
    const int st = c * D_STR;
    const _Float16* vp = Vt + ((size_t)g * DK_ + d) * S_LEN + st;
    for (int i = 0; i < L_CMP; ++i) {
      ks += (float)Kcr[((size_t)(st + i) * G_GRP + g) * DK_ + d];
      vs += (float)vp[i];
    }
    ks *= (1.0f / L_CMP);
    vs *= (1.0f / L_CMP);
  }
  Kcmp[((size_t)g * NCP_ + c) * DK_ + d]  = (_Float16)ks;
  Vcmpt[((size_t)g * DK_ + d) * NCP_ + c] = (_Float16)vs;
}

__global__ void zero_kernel(float* __restrict__ p, int n)
{
  const int i = blockIdx.x * blockDim.x + threadIdx.x;
  if (i < n) p[i] = 0.0f;
}

// ---------------------------------------------------------------------------
// Compressed attention + p_grp.  One wave per (16-row query tile, g, h).
// All operands f16, fragments are contiguous 16-half (32B) loads.
// ---------------------------------------------------------------------------
__global__ __launch_bounds__(32) void attn_cmp_kernel(
    const _Float16* __restrict__ Q, const _Float16* __restrict__ Kcmp,
    const _Float16* __restrict__ Vcmpt, float* __restrict__ Ocmp,
    float* __restrict__ pgrp)
{
  const int s0 = blockIdx.x * 16;
  const int g  = blockIdx.y / H_PER_G;
  const int h  = blockIdx.y % H_PER_G;
  const int lane  = threadIdx.x;
  const int r     = lane & 15;
  const int kbA   = (lane < 16) ? 0 : 16;
  const int kbB   = (lane < 16) ? 0 : 16;
  const int rbase = (lane < 16) ? 0 : 8;
  __shared__ _Float16 plds[16 * 128];
  __shared__ float    pg[16 * NB_];

  v16h qf[4];
  {
    const _Float16* qp = Q + (((size_t)(s0 + r) * G_GRP + g) * H_PER_G + h) * DK_;
#pragma unroll
    for (int kk = 0; kk < 4; ++kk)
#pragma unroll
      for (int i = 0; i < 8; ++i) {
        qf[kk][i]     = qp[kk * 32 + kbA + i];
        qf[kk][8 + i] = qp[kk * 32 + kbA + 8 + i];
      }
  }

  // scores: 8 tiles of 16 tokens
  v8f sc[8];
#pragma unroll
  for (int nt = 0; nt < 8; ++nt) {
#pragma unroll
    for (int i = 0; i < 8; ++i) sc[nt][i] = 0.0f;
    const _Float16* kp = Kcmp + ((size_t)g * NCP_ + nt * 16 + r) * DK_;
#pragma unroll
    for (int kk = 0; kk < 4; ++kk) {
      v16h bf;
#pragma unroll
      for (int i = 0; i < 16; ++i) bf[i] = kp[kk * 32 + kbB + i];
      sc[nt] = wmma16(qf[kk], bf, sc[nt]);
    }
  }

  // mask + softmax (rowwise over 127 tokens)
#pragma unroll
  for (int rr = 0; rr < 8; ++rr) {
    const int row = s0 + rbase + rr;
    float m = NEGL;
#pragma unroll
    for (int nt = 0; nt < 8; ++nt) {
      const int tok = nt * 16 + r;
      const bool ok = (tok < NC_) && (tok * D_STR + L_CMP - 1 <= row);
      const float v = ok ? sc[nt][rr] * SCALE_ : NEGL;
      sc[nt][rr] = v;
      m = fmaxf(m, v);
    }
    m = redmax16(m);
    float s = 0.0f;
#pragma unroll
    for (int nt = 0; nt < 8; ++nt) {
      const float p = (sc[nt][rr] <= -1e29f) ? 0.0f : __expf(sc[nt][rr] - m);
      sc[nt][rr] = p;
      s += p;
    }
    s = redsum16(s);
    const float invs = (s > 0.0f) ? (1.0f / s) : 0.0f;
#pragma unroll
    for (int nt = 0; nt < 8; ++nt) sc[nt][rr] *= invs;
  }

  // write P to LDS; accumulate p_grp into LDS
  for (int i = lane; i < 16 * NB_; i += 32) pg[i] = 0.0f;
  __syncthreads();
#pragma unroll
  for (int nt = 0; nt < 8; ++nt)
#pragma unroll
    for (int rr = 0; rr < 8; ++rr) {
      const float p = sc[nt][rr];
      plds[(rbase + rr) * 128 + nt * 16 + r] = (_Float16)p;
      const int tok = nt * 16 + r;
      if (p > 0.0f && tok < NC_) {
        const int a0 = tok * D_STR;                // compressed block start
        const int n1 = a0 >> 6;
        const int n2 = (a0 + L_CMP - 1) >> 6;
        const int ov1 = min(a0 + L_CMP, (n1 + 1) * 64) - a0;
        atomicAdd(&pg[(rbase + rr) * NB_ + n1], p * (float)ov1 * (1.0f / L_CMP));
        if (n2 != n1) {
          const int ov2 = (a0 + L_CMP) - n2 * 64;
          atomicAdd(&pg[(rbase + rr) * NB_ + n2], p * (float)ov2 * (1.0f / L_CMP));
        }
      }
    }
  __syncthreads();
  // flush p_grp (sum over h via global atomics)
  for (int i = lane; i < 16 * NB_; i += 32) {
    const int rr2 = i / NB_, n = i % NB_;
    atomicAdd(&pgrp[((size_t)(s0 + rr2) * G_GRP + g) * NB_ + n], pg[i]);
  }

  // O = P @ V_cmp   (V in transposed layout -> contiguous B fragments)
  v8f oacc[8];
#pragma unroll
  for (int nt = 0; nt < 8; ++nt)
#pragma unroll
    for (int i = 0; i < 8; ++i) oacc[nt][i] = 0.0f;
#pragma unroll
  for (int kk = 0; kk < 4; ++kk) {
    v16h ap;
#pragma unroll
    for (int i = 0; i < 8; ++i) {
      ap[i]     = plds[r * 128 + kk * 32 + kbA + i];
      ap[8 + i] = plds[r * 128 + kk * 32 + kbA + 8 + i];
    }
#pragma unroll
    for (int nt = 0; nt < 8; ++nt) {
      v16h bf;
      const _Float16* vp =
          Vcmpt + ((size_t)g * DK_ + nt * 16 + r) * NCP_ + kk * 32 + kbB;
#pragma unroll
      for (int i = 0; i < 16; ++i) bf[i] = vp[i];
      oacc[nt] = wmma16(ap, bf, oacc[nt]);
    }
  }
#pragma unroll
  for (int nt = 0; nt < 8; ++nt)
#pragma unroll
    for (int rr = 0; rr < 8; ++rr)
      Ocmp[(((size_t)(s0 + rbase + rr) * G_GRP + g) * H_PER_G + h) * DK_ + nt * 16 + r] =
          oacc[nt][rr];
}

// ---------------------------------------------------------------------------
// Top-16-of-32 block selection per (s,g), JAX top_k tie-breaking (stable).
// ---------------------------------------------------------------------------
__global__ void select_kernel(const float* __restrict__ pgrp,
                              unsigned* __restrict__ selmask)
{
  const int idx = blockIdx.x * blockDim.x + threadIdx.x;
  if (idx >= S_LEN * G_GRP) return;
  const int s = idx / G_GRP;
  float scr[NB_];
  const float* p = pgrp + (size_t)idx * NB_;
  const int cb = s / LSEL_;
  for (int n = 0; n < NB_; ++n) {
    float v = p[n];
    if (n == 0 || n == cb) v = 1e9f;
    else if (n * LSEL_ > s) v = -1e9f;
    scr[n] = v;
  }
  unsigned mask = 0;
  for (int it = 0; it < NSEL_; ++it) {
    float best = -INFINITY;
    int bi = 0;
    for (int n = 0; n < NB_; ++n)
      if (!((mask >> n) & 1u) && scr[n] > best) { best = scr[n]; bi = n; }
    mask |= (1u << bi);
  }
  selmask[idx] = mask;
}

// ---------------------------------------------------------------------------
// Flash attention.  K:(S,G,DK) f16, V transposed:(G,DK,S) f16.
// mode 0 = selected-blocks, 1 = sliding window.
// One wave per (16-row tile, g, h); online softmax, 32-key pairs.
// ---------------------------------------------------------------------------
__global__ __launch_bounds__(32) void attn_flash_kernel(
    const _Float16* __restrict__ Q, const _Float16* __restrict__ Km,
    const _Float16* __restrict__ Vt, const unsigned* __restrict__ selmask,
    float* __restrict__ Out, int mode)
{
  const int s0 = blockIdx.x * 16;
  const int g  = blockIdx.y / H_PER_G;
  const int h  = blockIdx.y % H_PER_G;
  const int lane  = threadIdx.x;
  const int r     = lane & 15;
  const int kbA   = (lane < 16) ? 0 : 16;
  const int kbB   = (lane < 16) ? 0 : 16;
  const int rbase = (lane < 16) ? 0 : 8;
  __shared__ _Float16 plds[16 * 32];

  v16h qf[4];
  {
    const _Float16* qp = Q + (((size_t)(s0 + r) * G_GRP + g) * H_PER_G + h) * DK_;
#pragma unroll
    for (int kk = 0; kk < 4; ++kk)
#pragma unroll
      for (int i = 0; i < 8; ++i) {
        qf[kk][i]     = qp[kk * 32 + kbA + i];
        qf[kk][8 + i] = qp[kk * 32 + kbA + 8 + i];
      }
  }

  unsigned rowsel[8];
  unsigned um = 0;
  if (mode == 0) {
    unsigned u = 0;
#pragma unroll
    for (int rr = 0; rr < 8; ++rr) {
      rowsel[rr] = selmask[(size_t)(s0 + rbase + rr) * G_GRP + g];
      u |= rowsel[rr];
    }
    // union across the two half-waves (rows 0-7 | rows 8-15)
    u |= (unsigned)__builtin_amdgcn_ds_swizzle((int)u, 0x1f | (16 << 10));
    um = u;
  }

  v8f oacc[8];
  float rmax[8], rsum[8];
#pragma unroll
  for (int nt = 0; nt < 8; ++nt)
#pragma unroll
    for (int i = 0; i < 8; ++i) oacc[nt][i] = 0.0f;
#pragma unroll
  for (int rr = 0; rr < 8; ++rr) { rmax[rr] = NEGL; rsum[rr] = 0.0f; }

  auto process32 = [&](int j0) {
    v8f sct[2];
#pragma unroll
    for (int sub = 0; sub < 2; ++sub) {
      v8f a;
#pragma unroll
      for (int i = 0; i < 8; ++i) a[i] = 0.0f;
      const _Float16* kp = Km + ((size_t)(j0 + sub * 16 + r) * G_GRP + g) * DK_;
#pragma unroll
      for (int kk = 0; kk < 4; ++kk) {
        v16h bf;
#pragma unroll
        for (int i = 0; i < 16; ++i) bf[i] = kp[kk * 32 + kbB + i];
        a = wmma16(qf[kk], bf, a);
      }
#pragma unroll
      for (int rr = 0; rr < 8; ++rr) {
        const int row = s0 + rbase + rr;
        const int key = j0 + sub * 16 + r;
        const bool ok = (mode == 0)
            ? ((key <= row) && ((rowsel[rr] >> (key >> 6)) & 1u))
            : ((key <= row) && (key > row - W_WIN));
        a[rr] = ok ? a[rr] * SCALE_ : NEGL;
      }
      sct[sub] = a;
    }
#pragma unroll
    for (int rr = 0; rr < 8; ++rr) {
      const float m  = redmax16(fmaxf(sct[0][rr], sct[1][rr]));
      const float nm = fmaxf(rmax[rr], m);
      const float sc_ = __expf(rmax[rr] - nm);
      const float p0 = (sct[0][rr] <= -1e29f) ? 0.0f : __expf(sct[0][rr] - nm);
      const float p1 = (sct[1][rr] <= -1e29f) ? 0.0f : __expf(sct[1][rr] - nm);
      sct[0][rr] = p0;
      sct[1][rr] = p1;
      const float ps = redsum16(p0 + p1);
      rsum[rr] = rsum[rr] * sc_ + ps;
      rmax[rr] = nm;
#pragma unroll
      for (int nt = 0; nt < 8; ++nt) oacc[nt][rr] *= sc_;
    }
#pragma unroll
    for (int rr = 0; rr < 8; ++rr) {
      plds[(rbase + rr) * 32 + r]      = (_Float16)sct[0][rr];
      plds[(rbase + rr) * 32 + 16 + r] = (_Float16)sct[1][rr];
    }
    __syncthreads();
    v16h ap;
#pragma unroll
    for (int i = 0; i < 8; ++i) {
      ap[i]     = plds[r * 32 + kbA + i];
      ap[8 + i] = plds[r * 32 + kbA + 8 + i];
    }
#pragma unroll
    for (int nt = 0; nt < 8; ++nt) {
      v16h bf;
      const _Float16* vp = Vt + ((size_t)g * DK_ + nt * 16 + r) * S_LEN + j0 + kbB;
#pragma unroll
      for (int i = 0; i < 16; ++i) bf[i] = vp[i];
      oacc[nt] = wmma16(ap, bf, oacc[nt]);
    }
    __syncthreads();
  };

  if (mode == 1) {
    int lo = s0 - (W_WIN - 1);
    if (lo < 0) lo = 0;
    for (int j0 = lo & ~31; j0 <= s0 + 15; j0 += 32) process32(j0);
  } else {
    const int nmax = (s0 + 15) >> 6;
    for (int n = 0; n <= nmax; ++n)
      if ((um >> n) & 1u) { process32(n * LSEL_); process32(n * LSEL_ + 32); }
  }

#pragma unroll
  for (int rr = 0; rr < 8; ++rr) {
    const float inv = (rsum[rr] > 0.0f) ? (1.0f / rsum[rr]) : 0.0f;
#pragma unroll
    for (int nt = 0; nt < 8; ++nt) oacc[nt][rr] *= inv;
  }
#pragma unroll
  for (int nt = 0; nt < 8; ++nt)
#pragma unroll
    for (int rr = 0; rr < 8; ++rr)
      Out[(((size_t)(s0 + rbase + rr) * G_GRP + g) * H_PER_G + h) * DK_ + nt * 16 + r] =
          oacc[nt][rr];
}

// ---------------------------------------------------------------------------
// Gating MLP: qg = mean_h Q -> silu MLP -> softmax / hard one-hot.
// ---------------------------------------------------------------------------
__global__ __launch_bounds__(64) void gate_kernel(
    const _Float16* __restrict__ Q, const float* __restrict__ fc1w,
    const float* __restrict__ fc1b, const float* __restrict__ fc2w,
    const float* __restrict__ fc2b, float* __restrict__ gates)
{
  const int idx = blockIdx.x * blockDim.x + threadIdx.x;
  if (idx >= S_LEN * G_GRP) return;
  const int s = idx / G_GRP, g = idx % G_GRP;
  const _Float16* qp = Q + ((size_t)s * G_GRP + g) * H_PER_G * DK_;
  float qg[DK_];
  for (int d = 0; d < DK_; ++d)
    qg[d] = 0.25f * ((float)qp[d] + (float)qp[DK_ + d] +
                     (float)qp[2 * DK_ + d] + (float)qp[3 * DK_ + d]);
  float gl[3] = { fc2b[0], fc2b[1], fc2b[2] };
  for (int j = 0; j < GH_; ++j) {
    float a = fc1b[j];
    for (int d = 0; d < DK_; ++d) a += qg[d] * fc1w[d * GH_ + j];
    const float si = a / (1.0f + __expf(-a));   // silu
    gl[0] += si * fc2w[j * 3 + 0];
    gl[1] += si * fc2w[j * 3 + 1];
    gl[2] += si * fc2w[j * 3 + 2];
  }
  // TAU = 1.0 -> no scaling
  const float m  = fmaxf(gl[0], fmaxf(gl[1], gl[2]));
  const float e0 = __expf(gl[0] - m), e1 = __expf(gl[1] - m), e2 = __expf(gl[2] - m);
  const float inv = 1.0f / (e0 + e1 + e2);
  float p0 = e0 * inv, p1 = e1 * inv, p2 = e2 * inv;
  const float mn = fminf(gl[0], fminf(gl[1], gl[2]));
  const float t2 = gl[0] + gl[1] + gl[2] - m - mn;  // second largest
  if (m - t2 > 50.0f) {
    int am = 0;
    float bv = gl[0];
    if (gl[1] > bv) { bv = gl[1]; am = 1; }
    if (gl[2] > bv) { bv = gl[2]; am = 2; }
    p0 = (am == 0) ? 1.0f : 0.0f;
    p1 = (am == 1) ? 1.0f : 0.0f;
    p2 = (am == 2) ? 1.0f : 0.0f;
  }
  gates[(size_t)idx * 3 + 0] = p0;
  gates[(size_t)idx * 3 + 1] = p1;
  gates[(size_t)idx * 3 + 2] = p2;
}

__global__ void combine_kernel(const float* __restrict__ Ocmp,
                               const float* __restrict__ Osel,
                               const float* __restrict__ Owin,
                               const float* __restrict__ gates,
                               float* __restrict__ Ocomb)
{
  const size_t idx = (size_t)blockIdx.x * blockDim.x + threadIdx.x;
  if (idx >= (size_t)S_LEN * Q_COLS) return;
  const size_t s = idx / Q_COLS;
  const int col = (int)(idx % Q_COLS);
  const int g = col / (H_PER_G * DK_);
  const float* gt = gates + ((size_t)s * G_GRP + g) * 3;
  Ocomb[idx] = gt[0] * Ocmp[idx] + gt[1] * Osel[idx] + gt[2] * Owin[idx];
}

// ---------------------------------------------------------------------------
extern "C" void kernel_launch(void* const* d_in, const int* in_sizes, int n_in,
                              void* d_out, int out_size, void* d_ws, size_t ws_size,
                              hipStream_t stream)
{
  (void)in_sizes; (void)n_in; (void)out_size; (void)ws_size;
  const float* x      = (const float*)d_in[0];
  const float* W_Q    = (const float*)d_in[1];
  const float* W_Ksel = (const float*)d_in[2];
  const float* W_Vsel = (const float*)d_in[3];
  const float* W_Kwin = (const float*)d_in[4];
  const float* W_Vwin = (const float*)d_in[5];
  const float* W_Kcmp = (const float*)d_in[6];
  const float* W_Vcmp = (const float*)d_in[7];
  const float* W_out  = (const float*)d_in[8];
  const float* fc1w   = (const float*)d_in[9];
  const float* fc1b   = (const float*)d_in[10];
  const float* fc2w   = (const float*)d_in[11];
  const float* fc2b   = (const float*)d_in[12];
  float* out = (float*)d_out;

  // ---- carve workspace (256B-aligned chunks) ----
  char* wsb = (char*)d_ws;
  auto carve = [&](size_t bytes) -> void* {
    void* p = (void*)wsb;
    wsb += (bytes + 255) & ~(size_t)255;
    return p;
  };
  _Float16* Q     = (_Float16*)carve((size_t)S_LEN * Q_COLS * 2);
  _Float16* Ksel  = (_Float16*)carve((size_t)S_LEN * KV_COLS * 2);
  _Float16* Kwin  = (_Float16*)carve((size_t)S_LEN * KV_COLS * 2);
  _Float16* Kcr   = (_Float16*)carve((size_t)S_LEN * KV_COLS * 2);
  _Float16* Vselt = (_Float16*)carve((size_t)S_LEN * KV_COLS * 2);  // (G,DK,S)
  _Float16* Vwint = (_Float16*)carve((size_t)S_LEN * KV_COLS * 2);  // (G,DK,S)
  _Float16* Vcrt  = (_Float16*)carve((size_t)S_LEN * KV_COLS * 2);  // (G,DK,S)
  _Float16* Kcmp  = (_Float16*)carve((size_t)G_GRP * NCP_ * DK_ * 2);
  _Float16* Vcmpt = (_Float16*)carve((size_t)G_GRP * NCP_ * DK_ * 2);
  float* Ocmp  = (float*)carve((size_t)S_LEN * Q_COLS * 4);
  float* Osel  = (float*)carve((size_t)S_LEN * Q_COLS * 4);
  float* Owin  = (float*)carve((size_t)S_LEN * Q_COLS * 4);
  float* Ocomb = (float*)carve((size_t)S_LEN * Q_COLS * 4);
  float* pgrp  = (float*)carve((size_t)S_LEN * G_GRP * NB_ * 4);
  float* gates = (float*)carve((size_t)S_LEN * G_GRP * 3 * 4);
  unsigned* selmask = (unsigned*)carve((size_t)S_LEN * G_GRP * 4);

  // 1. projections (WMMA GEMMs); K/Q stored f16, V stored f16 transposed
  gemm_f16_kernel<_Float16, false><<<dim3(Q_COLS / 64, S_LEN / 64), 128, 0, stream>>>(
      x, W_Q, Q, S_LEN, Q_COLS, DIM_);
  const dim3 gkv(KV_COLS / 64, S_LEN / 64);
  gemm_f16_kernel<_Float16, false><<<gkv, 128, 0, stream>>>(x, W_Ksel, Ksel, S_LEN, KV_COLS, DIM_);
  gemm_f16_kernel<_Float16, false><<<gkv, 128, 0, stream>>>(x, W_Kwin, Kwin, S_LEN, KV_COLS, DIM_);
  gemm_f16_kernel<_Float16, false><<<gkv, 128, 0, stream>>>(x, W_Kcmp, Kcr, S_LEN, KV_COLS, DIM_);
  gemm_f16_kernel<_Float16, true ><<<gkv, 128, 0, stream>>>(x, W_Vsel, Vselt, S_LEN, KV_COLS, DIM_);
  gemm_f16_kernel<_Float16, true ><<<gkv, 128, 0, stream>>>(x, W_Vwin, Vwint, S_LEN, KV_COLS, DIM_);
  gemm_f16_kernel<_Float16, true ><<<gkv, 128, 0, stream>>>(x, W_Vcmp, Vcrt, S_LEN, KV_COLS, DIM_);

  // 2. RoPE (f16 in place)
  rope_kernel<<<(S_LEN * NHEADS * 64) / 256, 256, 0, stream>>>(Q, NHEADS);
  rope_kernel<<<(S_LEN * G_GRP * 64) / 256, 256, 0, stream>>>(Ksel, G_GRP);
  rope_kernel<<<(S_LEN * G_GRP * 64) / 256, 256, 0, stream>>>(Kwin, G_GRP);
  rope_kernel<<<(S_LEN * G_GRP * 64) / 256, 256, 0, stream>>>(Kcr, G_GRP);

  // 3. compression
  compress_kernel<<<G_GRP * NCP_, 128, 0, stream>>>(Kcr, Vcrt, Kcmp, Vcmpt);

  // 4. compressed attention (+ p_grp)
  zero_kernel<<<(S_LEN * G_GRP * NB_ + 255) / 256, 256, 0, stream>>>(
      pgrp, S_LEN * G_GRP * NB_);
  const dim3 gattn(S_LEN / 16, G_GRP * H_PER_G);
  attn_cmp_kernel<<<gattn, 32, 0, stream>>>(Q, Kcmp, Vcmpt, Ocmp, pgrp);

  // 5. block selection
  select_kernel<<<(S_LEN * G_GRP + 255) / 256, 256, 0, stream>>>(pgrp, selmask);

  // 6. selected + window attention (flash)
  attn_flash_kernel<<<gattn, 32, 0, stream>>>(Q, Ksel, Vselt, selmask, Osel, 0);
  attn_flash_kernel<<<gattn, 32, 0, stream>>>(Q, Kwin, Vwint, selmask, Owin, 1);

  // 7. gating
  gate_kernel<<<(S_LEN * G_GRP + 63) / 64, 64, 0, stream>>>(
      Q, fc1w, fc1b, fc2w, fc2b, gates);

  // 8. gated combine
  combine_kernel<<<(int)(((size_t)S_LEN * Q_COLS + 255) / 256), 256, 0, stream>>>(
      Ocmp, Osel, Owin, gates, Ocomb);

  // 9. output projection (fp32 out)
  gemm_f16_kernel<float, false><<<dim3(DIM_ / 64, S_LEN / 64), 128, 0, stream>>>(
      Ocomb, W_out, out, S_LEN, DIM_, Q_COLS);
}
